// Attention_89223650607439
// MI455X (gfx1250) — compile-verified, matrix operand order checked by reading, not run on hardware
//
#include <hip/hip_runtime.h>
#include <hip/hip_bf16.h>

// ---------------------------------------------------------------------------
// Attention block for MI455X (gfx1250), bf16 WMMA everywhere.
//   x(f32) -> bf16 ; w* -> transposed bf16 [N,K]
//   gemm_bf16_nt : q_lin/k_lin/v_lin (f32)      (32x64 tile / wave, 8 wmma/k)
//   rope_q / rope_k / scatter_v -> head-major bf16 [B,H,T,HD]
//   flash_attn   : causal GQA flash attention; V staged to LDS with
//                  global_load_async_to_lds_b128 (ASYNCcnt), WMMA for S and PV
//   gemm_bf16_nt : out = y @ wo  (f32)
// ---------------------------------------------------------------------------

typedef __bf16 bf16;
typedef __attribute__((ext_vector_type(16))) __bf16 v16bf;
typedef __attribute__((ext_vector_type(8)))  __bf16 v8bf;
typedef __attribute__((ext_vector_type(8)))  float  v8f;

#define DEVINL __device__ __forceinline__

constexpr int Bc   = 2;
constexpr int Tc   = 2048;
constexpr int DIMc = 2048;
constexpr int NHc  = 32;
constexpr int NKVc = 8;
constexpr int HDc  = 64;
constexpr int REPc = NHc / NKVc;   // 4

// ---------------------------------------------------------------------------
// 16-bit WMMA operand loader (A: rows over lanes; B: cols over lanes).
// lanes 0-15 hold K {k0..k0+7} u {k0+16..k0+23},
// lanes 16-31 hold K {k0+8..k0+15} u {k0+24..k0+31}.
// ---------------------------------------------------------------------------
DEVINL v16bf load_tile16(const bf16* __restrict__ base, int ld, int k0, int lane) {
    const int half = lane >> 4;
    const int r    = lane & 15;
    const bf16* p  = base + (size_t)r * ld + k0 + half * 8;
    v8bf lo = *(const v8bf*)(p);
    v8bf hi = *(const v8bf*)(p + 16);
    v16bf out;
#pragma unroll
    for (int i = 0; i < 8; ++i) { out[i] = lo[i]; out[i + 8] = hi[i]; }
    return out;
}

DEVINL v8f wmma_bf16(v16bf a, v16bf b, v8f c) {
    // (neg_a, A, neg_b, B, c_mod, C, reuse_a, reuse_b)
    return __builtin_amdgcn_wmma_f32_16x16x32_bf16(false, a, false, b,
                                                   (short)0, c, false, false);
}

// CDNA5 async memory->LDS copy (tracked on ASYNCcnt). Per lane: 16 bytes.
DEVINL void async_copy_b128(unsigned lds_off, const bf16* gsrc) {
    asm volatile("global_load_async_to_lds_b128 %0, %1, off"
                 :: "v"(lds_off), "v"((unsigned long long)(uintptr_t)gsrc)
                 : "memory");
}
DEVINL void wait_asynccnt0() {
    asm volatile("s_wait_asynccnt 0x0" ::: "memory");
}
DEVINL void wait_dscnt0() {
    asm volatile("s_wait_dscnt 0x0" ::: "memory");
}

// ---------------------------------------------------------------------------
// C[M,N](f32) = A[M,K](bf16 row major) * Bt[N,K](bf16 row major == B^T)
// 128 threads = 4 waves, 2x2 wave grid; each wave owns a 32x64 tile
// (2x4 accumulators, 8 wmma per 32-wide K step) -> block tile 64x128.
// ---------------------------------------------------------------------------
__global__ __launch_bounds__(128)
void gemm_bf16_nt(const bf16* __restrict__ A, const bf16* __restrict__ Bt,
                  float* __restrict__ C, int M, int N, int K) {
    const int lane = threadIdx.x & 31;
    const int wave = threadIdx.x >> 5;
    const int wm   = wave >> 1;
    const int wn   = wave & 1;
    const int m0   = blockIdx.x * 64  + wm * 32;
    const int n0   = blockIdx.y * 128 + wn * 64;

    const bf16* Ar[2] = { A + (size_t)m0 * K, A + (size_t)(m0 + 16) * K };
    const bf16* Br[4] = { Bt + (size_t)n0 * K,        Bt + (size_t)(n0 + 16) * K,
                          Bt + (size_t)(n0 + 32) * K, Bt + (size_t)(n0 + 48) * K };

    v8f acc[2][4] = {};

    for (int k0 = 0; k0 < K; k0 += 32) {
        if (k0 + 64 < K) {  // global_prefetch_b8 of the next K chunk
            __builtin_prefetch(Ar[0] + (size_t)(lane & 15) * K + k0 + 64, 0, 3);
            __builtin_prefetch(Br[0] + (size_t)(lane & 15) * K + k0 + 64, 0, 3);
            __builtin_prefetch(Br[2] + (size_t)(lane & 15) * K + k0 + 64, 0, 3);
        }
        v16bf a0 = load_tile16(Ar[0], K, k0, lane);
        v16bf a1 = load_tile16(Ar[1], K, k0, lane);
        v16bf b0 = load_tile16(Br[0], K, k0, lane);
        v16bf b1 = load_tile16(Br[1], K, k0, lane);
        v16bf b2 = load_tile16(Br[2], K, k0, lane);
        v16bf b3 = load_tile16(Br[3], K, k0, lane);
        acc[0][0] = wmma_bf16(a0, b0, acc[0][0]);
        acc[0][1] = wmma_bf16(a0, b1, acc[0][1]);
        acc[0][2] = wmma_bf16(a0, b2, acc[0][2]);
        acc[0][3] = wmma_bf16(a0, b3, acc[0][3]);
        acc[1][0] = wmma_bf16(a1, b0, acc[1][0]);
        acc[1][1] = wmma_bf16(a1, b1, acc[1][1]);
        acc[1][2] = wmma_bf16(a1, b2, acc[1][2]);
        acc[1][3] = wmma_bf16(a1, b3, acc[1][3]);
    }

    const int half = lane >> 4;
    const int col  = lane & 15;
#pragma unroll
    for (int i = 0; i < 2; ++i)
#pragma unroll
        for (int j = 0; j < 4; ++j)
#pragma unroll
            for (int r = 0; r < 8; ++r)
                C[(size_t)(m0 + i * 16 + half * 8 + r) * N + (n0 + j * 16 + col)]
                    = acc[i][j][r];
}

// ---------------------------------------------------------------------------
// Elementwise helpers
// ---------------------------------------------------------------------------
__global__ void cvt_f32_to_bf16(const float* __restrict__ in,
                                bf16* __restrict__ out, size_t n) {
    size_t i = (size_t)blockIdx.x * blockDim.x + threadIdx.x;
    if (i < n) out[i] = (bf16)in[i];
}

// in: [K,N] f32 row major -> out: [N,K] bf16 row major
__global__ void transpose_to_bf16(const float* __restrict__ in,
                                  bf16* __restrict__ out, int K, int N) {
    size_t i = (size_t)blockIdx.x * blockDim.x + threadIdx.x;
    if (i >= (size_t)K * N) return;
    int k = (int)(i / N);
    int n = (int)(i % N);
    out[(size_t)n * K + k] = (bf16)in[i];
}

// q_lin [B*T, NH*HD] f32 -> qb [B,NH,T,HD] bf16 with RoPE and 1/sqrt(HD)
__global__ void rope_q(const float* __restrict__ qlin,
                       const float* __restrict__ fcos,
                       const float* __restrict__ fsin,
                       bf16* __restrict__ qb) {
    size_t i = (size_t)blockIdx.x * blockDim.x + threadIdx.x;
    const size_t total = (size_t)Bc * Tc * NHc * (HDc / 2);
    if (i >= total) return;
    int p = (int)(i % (HDc / 2)); size_t rem = i / (HDc / 2);
    int h = (int)(rem % NHc); rem /= NHc;
    int t = (int)(rem % Tc);
    int b = (int)(rem / Tc);
    const float* src = qlin + ((size_t)(b * Tc + t) * NHc + h) * HDc + 2 * p;
    float t0 = src[0], t1 = src[1];
    float c = fcos[(size_t)t * (HDc / 2) + p];
    float s = fsin[(size_t)t * (HDc / 2) + p];
    const float scale = 0.125f;  // 1/sqrt(64)
    bf16* dst = qb + ((size_t)(b * NHc + h) * Tc + t) * HDc + 2 * p;
    dst[0] = (bf16)((t0 * c - t1 * s) * scale);
    dst[1] = (bf16)((t0 * s + t1 * c) * scale);
}

// k_lin [B*T, NKV*HD] f32 -> kb [B,NKV,T,HD] bf16 with RoPE
__global__ void rope_k(const float* __restrict__ klin,
                       const float* __restrict__ fcos,
                       const float* __restrict__ fsin,
                       bf16* __restrict__ kb) {
    size_t i = (size_t)blockIdx.x * blockDim.x + threadIdx.x;
    const size_t total = (size_t)Bc * Tc * NKVc * (HDc / 2);
    if (i >= total) return;
    int p = (int)(i % (HDc / 2)); size_t rem = i / (HDc / 2);
    int g = (int)(rem % NKVc); rem /= NKVc;
    int t = (int)(rem % Tc);
    int b = (int)(rem / Tc);
    const float* src = klin + ((size_t)(b * Tc + t) * NKVc + g) * HDc + 2 * p;
    float t0 = src[0], t1 = src[1];
    float c = fcos[(size_t)t * (HDc / 2) + p];
    float s = fsin[(size_t)t * (HDc / 2) + p];
    bf16* dst = kb + ((size_t)(b * NKVc + g) * Tc + t) * HDc + 2 * p;
    dst[0] = (bf16)(t0 * c - t1 * s);
    dst[1] = (bf16)(t0 * s + t1 * c);
}

// v_lin [B*T, NKV*HD] f32 -> vb [B,NKV,T,HD] bf16
__global__ void scatter_v(const float* __restrict__ vlin, bf16* __restrict__ vb) {
    size_t i = (size_t)blockIdx.x * blockDim.x + threadIdx.x;
    const size_t total = (size_t)Bc * Tc * NKVc * HDc;
    if (i >= total) return;
    int d = (int)(i % HDc); size_t rem = i / HDc;
    int g = (int)(rem % NKVc); rem /= NKVc;
    int t = (int)(rem % Tc);
    int b = (int)(rem / Tc);
    vb[((size_t)(b * NKVc + g) * Tc + t) * HDc + d]
        = (bf16)vlin[((size_t)(b * Tc + t) * NKVc + g) * HDc + d];
}

// ---------------------------------------------------------------------------
// Causal GQA flash attention.
// One wave per 16-query tile; 4 independent waves per block, no cross-wave
// sync (each wave owns private LDS slabs for P and the V tile).
// V tile (32 keys x 64 dims, contiguous 4KB) is DMA'd into LDS with
// global_load_async_to_lds_b128, overlapping the S=QK^T WMMAs + softmax.
// C/D layout: lane half h, element i -> row = 8*h + i, col = lane&15.
// ---------------------------------------------------------------------------
__global__ __launch_bounds__(128)
void flash_attn(const bf16* __restrict__ qb, const bf16* __restrict__ kb,
                const bf16* __restrict__ vb, bf16* __restrict__ y) {
    __shared__ bf16 plds[4][16 * 32];      // P staging, per wave
    __shared__ bf16 vlds[4][32 * 64];      // V tile,    per wave

    const int lane = threadIdx.x & 31;
    const int wave = threadIdx.x >> 5;
    const int tiles_per_head = Tc / 16;            // 128
    const int gid = blockIdx.x * 4 + wave;         // over B*NH*tiles
    const int qt  = gid % tiles_per_head;
    const int bh  = gid / tiles_per_head;
    const int h   = bh % NHc;
    const int b   = bh / NHc;
    const int g   = h / REPc;
    const int q0  = qt * 16;

    const bf16* Q  = qb + (size_t)(b * NHc  + h) * Tc * HDc;
    const bf16* Kk = kb + (size_t)(b * NKVc + g) * Tc * HDc;
    const bf16* V  = vb + (size_t)(b * NKVc + g) * Tc * HDc;

    const v16bf aq0 = load_tile16(Q + (size_t)q0 * HDc, HDc, 0,  lane);
    const v16bf aq1 = load_tile16(Q + (size_t)q0 * HDc, HDc, 32, lane);

    v8f o[4] = {};
    float mrow[8], lrow[8];
#pragma unroll
    for (int i = 0; i < 8; ++i) { mrow[i] = -__builtin_inff(); lrow[i] = 0.f; }

    const int half = lane >> 4;
    const int col  = lane & 15;
    bf16* pbuf = &plds[wave][0];
    bf16* vt   = &vlds[wave][0];
    const unsigned vldsbase = (unsigned)(uintptr_t)vt;   // low 32 bits = LDS offset

    for (int kv0 = 0; kv0 < q0 + 16; kv0 += 32) {
        // make sure last iteration's LDS reads retired before overwriting V tile
        wait_dscnt0();
        // ---- kick async DMA of the contiguous V tile (32x64 bf16 = 4KB) ----
        {
            const bf16* vsrc = V + (size_t)kv0 * HDc;
#pragma unroll
            for (int j = 0; j < 8; ++j)
                async_copy_b128(vldsbase + (unsigned)(j * 512 + lane * 16),
                                vsrc + (j * 256 + lane * 8));
        }
        // ---- S = Q K^T over a 32-key block (two 16-key subtiles) ----
        v8f s0 = {}, s1 = {};
        {
            v16bf bk = load_tile16(Kk + (size_t)kv0 * HDc, HDc, 0, lane);
            s0 = wmma_bf16(aq0, bk, s0);
            bk = load_tile16(Kk + (size_t)kv0 * HDc, HDc, 32, lane);
            s0 = wmma_bf16(aq1, bk, s0);
            bk = load_tile16(Kk + (size_t)(kv0 + 16) * HDc, HDc, 0, lane);
            s1 = wmma_bf16(aq0, bk, s1);
            bk = load_tile16(Kk + (size_t)(kv0 + 16) * HDc, HDc, 32, lane);
            s1 = wmma_bf16(aq1, bk, s1);
        }
        // ---- causal mask + online softmax (per-row, 16-lane reductions) ----
#pragma unroll
        for (int i = 0; i < 8; ++i) {
            const int row = q0 + half * 8 + i;
            float v0 = (kv0 + col      <= row) ? s0[i] : -1e30f;
            float v1 = (kv0 + 16 + col <= row) ? s1[i] : -1e30f;
            float mx = fmaxf(v0, v1);
#pragma unroll
            for (int d = 1; d < 16; d <<= 1) mx = fmaxf(mx, __shfl_xor(mx, d, 32));
            const float mnew  = fmaxf(mrow[i], mx);
            const float p0    = __expf(v0 - mnew);
            const float p1    = __expf(v1 - mnew);
            float rs = p0 + p1;
#pragma unroll
            for (int d = 1; d < 16; d <<= 1) rs += __shfl_xor(rs, d, 32);
            const float alpha = __expf(mrow[i] - mnew);
            lrow[i] = lrow[i] * alpha + rs;
            mrow[i] = mnew;
#pragma unroll
            for (int n = 0; n < 4; ++n) o[n][i] *= alpha;
            // stage P (C layout -> row-major [16 x 32]) for reload in A layout
            pbuf[(half * 8 + i) * 32 + col]      = (bf16)p0;
            pbuf[(half * 8 + i) * 32 + 16 + col] = (bf16)p1;
        }
        // P: same-wave LDS RAW (DS in-order; fence compiler + wait)
        wait_dscnt0();
        // V tile DMA must have landed in LDS
        wait_asynccnt0();

        // ---- O += P V (A from P-LDS, B from V-LDS) ----
        const v16bf ap = load_tile16(pbuf, 32, 0, lane);
#pragma unroll
        for (int n = 0; n < 4; ++n) {
            v16bf bv;
            const int dim = n * 16 + col;
#pragma unroll
            for (int e = 0; e < 16; ++e) {
                const int key = (e < 8) ? (half * 8 + e) : (16 + half * 8 + (e - 8));
                bv[e] = vt[key * HDc + dim];
            }
            o[n] = wmma_bf16(ap, bv, o[n]);
        }
    }

    // ---- normalize and write y [B*T, NH*HD] bf16 ----
#pragma unroll
    for (int n = 0; n < 4; ++n)
#pragma unroll
        for (int i = 0; i < 8; ++i) {
            const float val = o[n][i] / lrow[i];
            y[(size_t)(b * Tc + q0 + half * 8 + i) * (NHc * HDc)
              + h * HDc + n * 16 + col] = (bf16)val;
        }
}

// ---------------------------------------------------------------------------
extern "C" void kernel_launch(void* const* d_in, const int* in_sizes, int n_in,
                              void* d_out, int out_size, void* d_ws, size_t ws_size,
                              hipStream_t stream) {
    (void)in_sizes; (void)n_in; (void)out_size; (void)ws_size;
    const float* x    = (const float*)d_in[0];
    const float* fcos = (const float*)d_in[1];
    const float* fsin = (const float*)d_in[2];
    const float* wq   = (const float*)d_in[3];
    const float* wk   = (const float*)d_in[4];
    const float* wv   = (const float*)d_in[5];
    const float* wo   = (const float*)d_in[6];
    float* out = (float*)d_out;

    char* ws = (char*)d_ws;
    size_t off = 0;
    auto alloc = [&](size_t bytes) {
        void* p = ws + off;
        off = (off + bytes + 255) & ~(size_t)255;
        return p;
    };

    const size_t M = (size_t)Bc * Tc;          // 4096
    bf16*  xb    = (bf16*) alloc(M * DIMc * 2);
    bf16*  wqt   = (bf16*) alloc((size_t)DIMc * (NHc * HDc) * 2);
    bf16*  wkt   = (bf16*) alloc((size_t)DIMc * (NKVc * HDc) * 2);
    bf16*  wvt   = (bf16*) alloc((size_t)DIMc * (NKVc * HDc) * 2);
    bf16*  wot   = (bf16*) alloc((size_t)(NHc * HDc) * DIMc * 2);
    float* q_lin = (float*)alloc(M * (NHc * HDc) * 4);
    float* k_lin = (float*)alloc(M * (NKVc * HDc) * 4);
    float* v_lin = (float*)alloc(M * (NKVc * HDc) * 4);
    bf16*  qb    = (bf16*) alloc(M * (NHc * HDc) * 2);
    bf16*  kbuf  = (bf16*) alloc(M * (NKVc * HDc) * 2);
    bf16*  vbuf  = (bf16*) alloc(M * (NKVc * HDc) * 2);
    bf16*  yb    = (bf16*) alloc(M * (NHc * HDc) * 2);

    const int TPB = 256;
    // 1) precision conversion / weight transposition
    {
        size_t n = M * DIMc;
        cvt_f32_to_bf16<<<dim3((unsigned)((n + TPB - 1) / TPB)), TPB, 0, stream>>>(x, xb, n);
    }
    transpose_to_bf16<<<dim3((DIMc * NHc * HDc) / TPB), TPB, 0, stream>>>(wq, wqt, DIMc, NHc * HDc);
    transpose_to_bf16<<<dim3((DIMc * NKVc * HDc) / TPB), TPB, 0, stream>>>(wk, wkt, DIMc, NKVc * HDc);
    transpose_to_bf16<<<dim3((DIMc * NKVc * HDc) / TPB), TPB, 0, stream>>>(wv, wvt, DIMc, NKVc * HDc);
    transpose_to_bf16<<<dim3(((NHc * HDc) * DIMc) / TPB), TPB, 0, stream>>>(wo, wot, NHc * HDc, DIMc);

    // 2) QKV projections (WMMA GEMMs): block tile 64x128
    gemm_bf16_nt<<<dim3((unsigned)(M / 64), (NHc * HDc) / 128),  128, 0, stream>>>(xb, wqt, q_lin, (int)M, NHc * HDc,  DIMc);
    gemm_bf16_nt<<<dim3((unsigned)(M / 64), (NKVc * HDc) / 128), 128, 0, stream>>>(xb, wkt, k_lin, (int)M, NKVc * HDc, DIMc);
    gemm_bf16_nt<<<dim3((unsigned)(M / 64), (NKVc * HDc) / 128), 128, 0, stream>>>(xb, wvt, v_lin, (int)M, NKVc * HDc, DIMc);

    // 3) RoPE + head-major layout
    rope_q<<<dim3((unsigned)((M * NHc * (HDc / 2)) / TPB)), TPB, 0, stream>>>(q_lin, fcos, fsin, qb);
    rope_k<<<dim3((unsigned)((M * NKVc * (HDc / 2)) / TPB)), TPB, 0, stream>>>(k_lin, fcos, fsin, kbuf);
    scatter_v<<<dim3((unsigned)((M * NKVc * HDc) / TPB)), TPB, 0, stream>>>(v_lin, vbuf);

    // 4) flash attention: B*NH*(T/16) tiles / 4 waves per block
    {
        const unsigned nblk = (unsigned)((size_t)Bc * NHc * (Tc / 16) / 4);  // 2048
        flash_attn<<<dim3(nblk), 128, 0, stream>>>(qb, kbuf, vbuf, yb);
    }

    // 5) output projection
    gemm_bf16_nt<<<dim3((unsigned)(M / 64), DIMc / 128), 128, 0, stream>>>(yb, wot, out, (int)M, DIMc, NHc * HDc);
}
